// SparseGLU_Llama_62113817035353
// MI455X (gfx1250) — compile-verified, matrix operand order checked by reading, not run on hardware
//
#include <hip/hip_runtime.h>
#include <hip/hip_bf16.h>

typedef __bf16 bf16;
typedef __attribute__((ext_vector_type(16))) __bf16 v16bf;
typedef __attribute__((ext_vector_type(8)))  __bf16 v8bf;
typedef __attribute__((ext_vector_type(4)))  __bf16 v4bf;
typedef __attribute__((ext_vector_type(8)))  float  v8f;

// Problem constants (from reference): E experts, T tokens, H hidden, F ffn
#define E_   8
#define T_   8192
#define H_   4096
#define F_   2048
#define TPE_ (T_ / E_)          // 1024 tokens per expert

// Tiling
#define TM 128                  // tokens per block tile
#define TN 128                  // output cols per block tile
#define TK 64                   // K step (2 x WMMA K=32)
#define LDS_STRIDE (TK + 8)     // 72 bf16 = 144B rows: bank spread, keeps 16B align

// ---------------------------------------------------------------------------
// CDNA5 async copy: global -> LDS 128-bit, tracked on ASYNCcnt (no VGPR data).
// LDS operand is the low 32 bits of the generic shared pointer (ISA 10.2:
// flat->LDS address = addr[31:0]); HW adds the wave's LDS_BASE.
// ---------------------------------------------------------------------------
__device__ __forceinline__ void async_copy_b128(const void* gsrc, void* ldst) {
    unsigned lds_off = (unsigned)(unsigned long long)ldst;
    asm volatile("global_load_async_to_lds_b128 %0, %1, off"
                 :: "v"(lds_off), "v"(gsrc) : "memory");
}

__device__ __forceinline__ void wait_async0() {
#if __has_builtin(__builtin_amdgcn_s_wait_asynccnt)
    __builtin_amdgcn_s_wait_asynccnt(0);
#else
    asm volatile("s_wait_asynccnt 0" ::: "memory");
#endif
}

// ---------------------------------------------------------------------------
// Load a 16x32 bf16 WMMA fragment (A or B operand) from an LDS tile stored
// [row][k] with stride LDS_STRIDE. Layout per CDNA5 ISA (16-bit A 16x32):
//   lane half h = lane>>4 ; element i -> K = (i<8?0:16) + h*8 + (i&7)
// ---------------------------------------------------------------------------
__device__ __forceinline__ v16bf load_frag(const bf16* lds, int row, int k0, int h8) {
    const bf16* p = lds + row * LDS_STRIDE + k0 + h8;
    v8bf lo = *(const v8bf*)(p);        // K = k0+h8   .. +7
    v8bf hi = *(const v8bf*)(p + 16);   // K = k0+h8+16.. +23
    return __builtin_shufflevector(lo, hi, 0,1,2,3,4,5,6,7,8,9,10,11,12,13,14,15);
}

// Load a TM x TK fp32 tile from global (row stride src_ld), convert to bf16,
// write to LDS [row][k]. 256 threads, float4 per iteration.
__device__ __forceinline__ void load_tile_f32(bf16* lds, const float* __restrict__ src,
                                              int src_ld, int tid) {
#pragma unroll
    for (int it = 0; it < (TM * TK / 4) / 256; ++it) {
        int idx = it * 256 + tid;
        int row = idx / (TK / 4);
        int c4  = (idx % (TK / 4)) * 4;
        const float4 v = *(const float4*)(src + (size_t)row * src_ld + c4);
        v4bf o;
        o[0] = (bf16)v.x; o[1] = (bf16)v.y; o[2] = (bf16)v.z; o[3] = (bf16)v.w;
        *(v4bf*)(lds + row * LDS_STRIDE + c4) = o;
    }
}

// ===========================================================================
// Kernel 1: per-expert  h1 = x*gate^T, h2 = x*up^T, hid = silu(h1)*h2 (bf16)
// grid: x = (TPE/TM)*(F/TN) = 8*16 = 128 tiles, y = expert
// ===========================================================================
__global__ __launch_bounds__(256) void moe_sdd_glu(const float* __restrict__ x,
                                                   const float* __restrict__ w1,
                                                   bf16* __restrict__ hid) {
    __shared__ __attribute__((aligned(16))) bf16 lA[TM * LDS_STRIDE];
    __shared__ __attribute__((aligned(16))) bf16 lG[TN * LDS_STRIDE];
    __shared__ __attribute__((aligned(16))) bf16 lU[TN * LDS_STRIDE];

    const int e   = blockIdx.y;
    const int tm  = blockIdx.x >> 4;     // 0..7
    const int tn  = blockIdx.x & 15;     // 0..15
    const int tid = threadIdx.x;
    const int lane = tid & 31;
    const int wid  = tid >> 5;           // 8 waves
    const int wm   = wid >> 1;           // 0..3 : 32-row slab
    const int wn   = wid & 1;            // 0..1 : 64-col slab
    const int h    = lane >> 4;
    const int ln   = lane & 15;

    const float* xb = x  + (size_t)(e * TPE_ + tm * TM) * H_;
    const float* gb = w1 + (size_t)(e * F_   + tn * TN) * H_;
    const float* ub = gb + (size_t)E_ * F_ * H_;

    // prefetch address pattern: 128 rows x 2 cachelines of the next K-tile
    const size_t pf_off = (size_t)(tid >> 1) * H_ + (tid & 1) * 32;

    v8f accg[2][4] = {};
    v8f accu[2][4] = {};

    for (int k0 = 0; k0 < H_; k0 += TK) {
        load_tile_f32(lA, xb + k0, H_, tid);
        load_tile_f32(lG, gb + k0, H_, tid);
        load_tile_f32(lU, ub + k0, H_, tid);
        if (k0 + TK < H_) {              // hint the next K-tile into cache
            __builtin_prefetch(xb + k0 + TK + pf_off, 0, 0);
            __builtin_prefetch(gb + k0 + TK + pf_off, 0, 0);
            __builtin_prefetch(ub + k0 + TK + pf_off, 0, 0);
        }
        __syncthreads();
#pragma unroll
        for (int kk = 0; kk < TK; kk += 32) {
            v16bf a[2];
#pragma unroll
            for (int mt = 0; mt < 2; ++mt)
                a[mt] = load_frag(lA, wm * 32 + mt * 16 + ln, kk, h * 8);
#pragma unroll
            for (int nt = 0; nt < 4; ++nt) {
                // load BOTH fragments first so one LDS wait covers 4 WMMAs
                v16bf bg = load_frag(lG, wn * 64 + nt * 16 + ln, kk, h * 8);
                v16bf bu = load_frag(lU, wn * 64 + nt * 16 + ln, kk, h * 8);
                accg[0][nt] = __builtin_amdgcn_wmma_f32_16x16x32_bf16(
                    false, a[0], false, bg, (short)0, accg[0][nt], false, false);
                accg[1][nt] = __builtin_amdgcn_wmma_f32_16x16x32_bf16(
                    false, a[1], false, bg, (short)0, accg[1][nt], false, false);
                accu[0][nt] = __builtin_amdgcn_wmma_f32_16x16x32_bf16(
                    false, a[0], false, bu, (short)0, accu[0][nt], false, false);
                accu[1][nt] = __builtin_amdgcn_wmma_f32_16x16x32_bf16(
                    false, a[1], false, bu, (short)0, accu[1][nt], false, false);
            }
        }
        __syncthreads();
    }

    // Fused swish-GLU epilogue, store bf16 hid to workspace.
    bf16* hb = hid + (size_t)(e * TPE_ + tm * TM) * F_ + tn * TN;
#pragma unroll
    for (int mt = 0; mt < 2; ++mt)
#pragma unroll
        for (int nt = 0; nt < 4; ++nt) {
            const int col = wn * 64 + nt * 16 + ln;
#pragma unroll
            for (int j = 0; j < 8; ++j) {
                const int row = wm * 32 + mt * 16 + h * 8 + j;
                const float g = accg[mt][nt][j];
                const float u = accu[mt][nt][j];
                const float s = g / (1.0f + __expf(-g));   // silu(g)
                hb[(size_t)row * F_ + col] = (bf16)(s * u);
            }
        }
}

// ===========================================================================
// Kernel 2: per-expert  out = hid * w2   ([Tpe,F] x [F,H] -> [Tpe,H], fp32)
// grid: x = (TPE/TM)*(H/TN) = 8*32 = 256 tiles, y = expert
// A tile (bf16) staged via CDNA5 async global->LDS copies (ASYNCcnt),
// overlapping the fp32->bf16 transpose of the B tile.
// ===========================================================================
__global__ __launch_bounds__(256) void moe_dsd(const bf16* __restrict__ hid,
                                               const float* __restrict__ w2,
                                               float* __restrict__ out) {
    __shared__ __attribute__((aligned(16))) bf16 lA[TM * LDS_STRIDE];
    __shared__ __attribute__((aligned(16))) bf16 lB[TN * LDS_STRIDE];   // [n][k]

    const int e   = blockIdx.y;
    const int tm  = blockIdx.x >> 5;     // 0..7
    const int tn  = blockIdx.x & 31;     // 0..31
    const int tid = threadIdx.x;
    const int lane = tid & 31;
    const int wid  = tid >> 5;
    const int wm   = wid >> 1;
    const int wn   = wid & 1;
    const int h    = lane >> 4;
    const int ln   = lane & 15;

    const bf16*  ab = hid + (size_t)(e * TPE_ + tm * TM) * F_;
    const float* bb = w2  + (size_t)e * F_ * H_ + tn * TN;   // rows = k (f), cols = n (h)

    // prefetch pattern for next w2 K-tile: 64 rows x 4 cachelines
    const size_t pf_off = (size_t)(tid >> 2) * H_ + (tid & 3) * 32;

    v8f acc[2][4] = {};

    for (int k0 = 0; k0 < F_; k0 += TK) {
        // A: async bf16 copy, TM x TK, [row][k]; 16B per lane per sweep
#pragma unroll
        for (int it = 0; it < (TM * TK / 8) / 256; ++it) {
            int idx = it * 256 + tid;
            int row = idx / (TK / 8);
            int c8  = (idx % (TK / 8)) * 8;
            async_copy_b128(ab + (size_t)row * F_ + k0 + c8,
                            lA + row * LDS_STRIDE + c8);
        }
        // B: fp32 -> bf16 transpose. Each thread: 4 coalesced global loads
        // (one column of 4 k-values; lanes span contiguous n) + 1 ds_store_b64.
#pragma unroll
        for (int it = 0; it < (TK * TN / 4) / 256; ++it) {
            int idx = it * 256 + tid;
            int n   = idx % TN;
            int k4  = (idx / TN) * 4;
            const float* p = bb + (size_t)(k0 + k4) * H_ + n;
            float a0 = p[0];
            float a1 = p[(size_t)H_];
            float a2 = p[(size_t)2 * H_];
            float a3 = p[(size_t)3 * H_];
            v4bf o;
            o[0] = (bf16)a0; o[1] = (bf16)a1; o[2] = (bf16)a2; o[3] = (bf16)a3;
            *(v4bf*)(lB + n * LDS_STRIDE + k4) = o;
        }
        if (k0 + TK < F_)
            __builtin_prefetch(bb + (size_t)(k0 + TK) * H_ + pf_off, 0, 0);

        wait_async0();                   // this wave's async A copies done
        __syncthreads();
#pragma unroll
        for (int kk = 0; kk < TK; kk += 32) {
            v16bf a[2];
#pragma unroll
            for (int mt = 0; mt < 2; ++mt)
                a[mt] = load_frag(lA, wm * 32 + mt * 16 + ln, kk, h * 8);
#pragma unroll
            for (int nt = 0; nt < 4; ++nt) {
                v16bf b = load_frag(lB, wn * 64 + nt * 16 + ln, kk, h * 8);
#pragma unroll
                for (int mt = 0; mt < 2; ++mt)
                    acc[mt][nt] = __builtin_amdgcn_wmma_f32_16x16x32_bf16(
                        false, a[mt], false, b, (short)0, acc[mt][nt], false, false);
            }
        }
        __syncthreads();
    }

    float* ob = out + (size_t)(e * TPE_ + tm * TM) * H_ + tn * TN;
#pragma unroll
    for (int mt = 0; mt < 2; ++mt)
#pragma unroll
        for (int nt = 0; nt < 4; ++nt) {
            const int col = wn * 64 + nt * 16 + ln;
#pragma unroll
            for (int j = 0; j < 8; ++j) {
                const int row = wm * 32 + mt * 16 + h * 8 + j;
                ob[(size_t)row * H_ + col] = acc[mt][nt][j];
            }
        }
}

// ===========================================================================
extern "C" void kernel_launch(void* const* d_in, const int* in_sizes, int n_in,
                              void* d_out, int out_size, void* d_ws, size_t ws_size,
                              hipStream_t stream) {
    const float* x  = (const float*)d_in[0];   // [T, H]
    const float* w1 = (const float*)d_in[1];   // [2*E*F, H]
    const float* w2 = (const float*)d_in[2];   // [E*F, H]
    // d_in[3] = num_experts (compile-time constant E_ here)

    bf16*  hid = (bf16*)d_ws;                  // [T, F] bf16 = 32 MB scratch
    float* out = (float*)d_out;                // [T, H] fp32

    dim3 blk(256);
    dim3 g1((TPE_ / TM) * (F_ / TN), E_);      // 128 x 8
    dim3 g2((TPE_ / TM) * (H_ / TN), E_);      // 256 x 8
    moe_sdd_glu<<<g1, blk, 0, stream>>>(x, w1, hid);
    moe_dsd   <<<g2, blk, 0, stream>>>(hid, w2, out);
}
